// SelfMultiHeadAttn_3143916060827
// MI455X (gfx1250) — matrix operand order repaired and checked by reading, not compiled, not run on hardware
//
#include <hip/hip_runtime.h>
#include <math.h>

// ---------- CDNA5 WMMA types ----------
typedef __attribute__((ext_vector_type(16))) __bf16 v16bf;
typedef __attribute__((ext_vector_type(8)))  __bf16 v8bf;
typedef __attribute__((ext_vector_type(8)))  float  v8f;

#define LDK 40   // bf16 LDS row stride: 80B, 16B-aligned, bank-friendly

__device__ __forceinline__ v8f wmma_bf16(v16bf a, v16bf b, v8f c) {
  // D = A(16x32 bf16) * B(32x16 bf16) + C(16x16 f32)
  return __builtin_amdgcn_wmma_f32_16x16x32_bf16(
      false, a, false, b, (short)0, c, false, false);
}

__device__ __forceinline__ v16bf combine(v8bf lo, v8bf hi) {
  v16bf r;
#pragma unroll
  for (int j = 0; j < 8; ++j) { r[j] = lo[j]; r[j + 8] = hi[j]; }
  return r;
}

// A fragment (16x32, MxK) from LDS row-major tile, stride LDK.
// ISA: lanes 0-15 hold K=0..7,16..23 ; lanes 16-31 hold K=8..15,24..31.
__device__ __forceinline__ v16bf load_a_lds(const __bf16* tile, int lane) {
  const int row = lane & 15, g = lane >> 4;
  const __bf16* p = tile + row * LDK;
  return combine(*(const v8bf*)(p + g * 8), *(const v8bf*)(p + 16 + g * 8));
}

// B fragment (32x16, KxN) from LDS [n][k] tile, stride LDK.
// ISA: lanes 0-15: K=0..7,16..23 for col n=lane ; lanes 16-31: K=8..15,24..31.
__device__ __forceinline__ v16bf load_b_lds(const __bf16* tile, int lane) {
  const int n = lane & 15, g = lane >> 4;
  const __bf16* p = tile + n * LDK + g * 8;
  return combine(*(const v8bf*)p, *(const v8bf*)(p + 16));
}

// ================= Kernel 1/4: bf16-WMMA GEMM (fp32 in/out) =================
// C[4096 x NCOLS] = A[4096 x 1024] * W[1024 x NCOLS] + bias (+ resid)
// Block: 128x128 tile, 8 waves; wave = 32x64 sub-tile = 2x4 WMMA accumulators.
template <int NCOLS, bool RES>
__global__ __launch_bounds__(256) void gemm_bf16_kernel(
    const float* __restrict__ A, const float* __restrict__ W,
    const float* __restrict__ bias, const float* __restrict__ resid,
    float* __restrict__ out, int Kdim) {
  __shared__ __bf16 sA[128 * LDK];
  __shared__ __bf16 sB[128 * LDK];
  const int tid = threadIdx.x;
  const int lane = tid & 31, wv = tid >> 5;
  const int row0 = blockIdx.y * 128, col0 = blockIdx.x * 128;
  const int wm = wv & 3, wn = wv >> 2;

  v8f acc[2][4];
#pragma unroll
  for (int a = 0; a < 2; ++a)
#pragma unroll
    for (int b = 0; b < 4; ++b) acc[a][b] = (v8f){};

  for (int k0 = 0; k0 < Kdim; k0 += 32) {
    // stage A (128x32) and B (32x128, transposed to [n][k]) as bf16
#pragma unroll
    for (int e = 0; e < 16; ++e) {
      const int idx = e * 256 + tid;
      const int r = idx >> 5, k = idx & 31;          // coalesced over k
      sA[r * LDK + k] = (__bf16)A[(size_t)(row0 + r) * Kdim + k0 + k];
      const int c = idx & 127, kr = idx >> 7;        // coalesced over cols
      sB[c * LDK + kr] = (__bf16)W[(size_t)(k0 + kr) * NCOLS + col0 + c];
    }
    __syncthreads();
    v16bf af[2], bfr[4];
#pragma unroll
    for (int tm = 0; tm < 2; ++tm)
      af[tm] = load_a_lds(&sA[(wm * 32 + tm * 16) * LDK], lane);
#pragma unroll
    for (int tn = 0; tn < 4; ++tn)
      bfr[tn] = load_b_lds(&sB[(wn * 64 + tn * 16) * LDK], lane);
#pragma unroll
    for (int tm = 0; tm < 2; ++tm)
#pragma unroll
      for (int tn = 0; tn < 4; ++tn)
        acc[tm][tn] = wmma_bf16(af[tm], bfr[tn], acc[tm][tn]);
    __syncthreads();
  }

  // epilogue: C layout is m = i + 8*g, n = lane&15
  const int n = lane & 15, g = lane >> 4;
#pragma unroll
  for (int tm = 0; tm < 2; ++tm) {
#pragma unroll
    for (int tn = 0; tn < 4; ++tn) {
      const int col = col0 + wn * 64 + tn * 16 + n;
      const float bv = bias[col];
#pragma unroll
      for (int i = 0; i < 8; ++i) {
        const int row = row0 + wm * 32 + tm * 16 + i + 8 * g;
        float v = acc[tm][tn][i] + bv;
        if (RES) v += resid[(size_t)row * NCOLS + col];
        out[(size_t)row * NCOLS + col] = v;
      }
    }
  }
}

// ============ Kernel 2: k-norm + RoPE + pack to bf16 (q,k row / v^T) ========
// one wave per (b,s,head); lane = half-dim index j in [0,32)
__global__ __launch_bounds__(256) void rope_pack_kernel(
    const float* __restrict__ qkv, const float* __restrict__ pos,
    __bf16* __restrict__ qo, __bf16* __restrict__ ko, __bf16* __restrict__ vt) {
  const int lane = threadIdx.x & 31;
  const int widx = blockIdx.x * 8 + (threadIdx.x >> 5);  // b*S*R + s*R + r
  const int r = widx & 15;
  const int s = (widx >> 4) & 2047;
  const int b = widx >> 15;
  const int row = b * 2048 + s;
  const float* base = qkv + (size_t)row * 3072 + r * 192;

  float q0 = base[lane],        q1 = base[lane + 32];
  float k0 = base[64 + lane],   k1 = base[96 + lane];
  float v0 = base[128 + lane],  v1 = base[160 + lane];

  // L2-normalize k over head dim (64 values = 2 per lane over 32 lanes)
  float ss = k0 * k0 + k1 * k1;
  ss += __shfl_xor(ss, 1);  ss += __shfl_xor(ss, 2);  ss += __shfl_xor(ss, 4);
  ss += __shfl_xor(ss, 8);  ss += __shfl_xor(ss, 16);
  const float rn = rsqrtf(ss);
  k0 *= rn; k1 *= rn;

  // RoPE: pos[...,:32]=sin, pos[...,32:]=cos
  const float* pp = pos + (size_t)row * 64;
  const float sn = pp[lane], cs = pp[lane + 32];
  const float qa = q0 * cs - q1 * sn, qb = q0 * sn + q1 * cs;
  const float ka = k0 * cs - k1 * sn, kb = k0 * sn + k1 * cs;

  const int hr = b * 16 + r;
  const size_t o = ((size_t)hr * 2048 + s) * 64;
  qo[o + lane] = (__bf16)qa;  qo[o + lane + 32] = (__bf16)qb;
  ko[o + lane] = (__bf16)ka;  ko[o + lane + 32] = (__bf16)kb;
  const size_t vb = (size_t)hr * 64 * 2048;              // v^T: [64][S]
  vt[vb + (size_t)lane * 2048 + s]        = (__bf16)v0;
  vt[vb + (size_t)(lane + 32) * 2048 + s] = (__bf16)v1;
}

// ================= Kernel 3: flash attention (WMMA bf16) ====================
// one wave per 16-query tile of one (b,head); online softmax over 2048 keys.
__global__ __launch_bounds__(256) void attn_kernel(
    const __bf16* __restrict__ qb, const __bf16* __restrict__ kb,
    const __bf16* __restrict__ vt, float* __restrict__ ctx) {
  __shared__ __bf16 pbuf[8][16 * LDK];   // per-wave P-tile bounce buffer
  const int lane = threadIdx.x & 31;
  const int wv = threadIdx.x >> 5;
  const int widx = blockIdx.x * 8 + wv;  // hr*(S/16) + qtile
  const int qt = widx & 127;
  const int hr = widx >> 7;
  const int b = hr >> 4, r = hr & 15;
  const int q0 = qt * 16;
  const __bf16* qh = qb + (size_t)hr * 2048 * 64;
  const __bf16* kh = kb + (size_t)hr * 2048 * 64;
  const __bf16* vh = vt + (size_t)hr * 64 * 2048;
  const int nn = lane & 15, g = lane >> 4;

  // Q as two A-fragments (head-dim 0..31 / 32..63), loaded once
  v16bf qa0, qa1;
  {
    const __bf16* p = qh + (size_t)(q0 + nn) * 64;
    qa0 = combine(*(const v8bf*)(p + g * 8),      *(const v8bf*)(p + 16 + g * 8));
    qa1 = combine(*(const v8bf*)(p + 32 + g * 8), *(const v8bf*)(p + 48 + g * 8));
  }

  v8f acc[4];
#pragma unroll
  for (int t = 0; t < 4; ++t) acc[t] = (v8f){};
  float mrow[8], lrow[8];
#pragma unroll
  for (int i = 0; i < 8; ++i) { mrow[i] = -3.0e38f; lrow[i] = 0.f; }

  __bf16* pb = pbuf[wv];

  for (int kc = 0; kc < 2048; kc += 32) {
    // ---- scores: two 16x16 tiles over 32 keys, K-dim = 64 head dims ----
    v8f sc[2];
#pragma unroll
    for (int t = 0; t < 2; ++t) {
      const __bf16* kp = kh + (size_t)(kc + t * 16 + nn) * 64;  // n=key, k=dim
      v16bf kf0 = combine(*(const v8bf*)(kp + g * 8),
                          *(const v8bf*)(kp + g * 8 + 16));
      v16bf kf1 = combine(*(const v8bf*)(kp + 32 + g * 8),
                          *(const v8bf*)(kp + 32 + g * 8 + 16));
      v8f z = (v8f){};
      sc[t] = wmma_bf16(qa0, kf0, z);
      sc[t] = wmma_bf16(qa1, kf1, sc[t]);
    }
    // ---- online softmax; row m = i + 8*g, cols across 16 lanes of half ----
    float alpha[8];
#pragma unroll
    for (int i = 0; i < 8; ++i) {
      float t0 = fmaxf(sc[0][i], sc[1][i]);
      t0 = fmaxf(t0, __shfl_xor(t0, 1));
      t0 = fmaxf(t0, __shfl_xor(t0, 2));
      t0 = fmaxf(t0, __shfl_xor(t0, 4));
      t0 = fmaxf(t0, __shfl_xor(t0, 8));
      const float mn = fmaxf(mrow[i], t0);
      alpha[i] = __expf(mrow[i] - mn);
      mrow[i] = mn;
      const float p0 = __expf(sc[0][i] - mn);
      const float p1 = __expf(sc[1][i] - mn);
      sc[0][i] = p0; sc[1][i] = p1;
      float rs = p0 + p1;
      rs += __shfl_xor(rs, 1); rs += __shfl_xor(rs, 2);
      rs += __shfl_xor(rs, 4); rs += __shfl_xor(rs, 8);
      lrow[i] = lrow[i] * alpha[i] + rs;
    }
#pragma unroll
    for (int t = 0; t < 4; ++t)
#pragma unroll
      for (int i = 0; i < 8; ++i) acc[t][i] *= alpha[i];

    // ---- re-shape P: C-layout -> LDS -> A-fragment (bf16) ----
#pragma unroll
    for (int i = 0; i < 8; ++i) {
      pb[(i + 8 * g) * LDK + nn]      = (__bf16)sc[0][i];
      pb[(i + 8 * g) * LDK + 16 + nn] = (__bf16)sc[1][i];
    }
    asm volatile("s_wait_dscnt 0x0" ::: "memory");  // same-wave LDS RAW
    v16bf pa;
    {
      const __bf16* p = pb + nn * LDK;
      pa = combine(*(const v8bf*)(p + g * 8), *(const v8bf*)(p + 16 + g * 8));
    }
    // ---- PV: A = P(16x32 keys), B = V^T slices (32 keys x 16 dims) ----
#pragma unroll
    for (int t = 0; t < 4; ++t) {
      const __bf16* vp = vh + (size_t)(t * 16 + nn) * 2048 + kc + g * 8;
      v16bf vf = combine(*(const v8bf*)vp, *(const v8bf*)(vp + 16));
      acc[t] = wmma_bf16(pa, vf, acc[t]);
    }
  }

  // epilogue: ctx[b, q, r*64 + d]
#pragma unroll
  for (int t = 0; t < 4; ++t)
#pragma unroll
    for (int i = 0; i < 8; ++i) {
      const int row = b * 2048 + q0 + i + 8 * g;
      const int col = r * 64 + t * 16 + nn;
      ctx[(size_t)row * 1024 + col] = acc[t][i] / lrow[i];
    }
}

// ====================== Kernel 5: row-wise LayerNorm ========================
__global__ __launch_bounds__(256) void ln_kernel(
    const float* __restrict__ y, const float* __restrict__ gam,
    const float* __restrict__ bet, float* __restrict__ out) {
  __shared__ float red[256];
  const int row = blockIdx.x, tid = threadIdx.x;
  const float* yr = y + (size_t)row * 1024;
  float v[4]; float s = 0.f;
#pragma unroll
  for (int e = 0; e < 4; ++e) { v[e] = yr[tid + e * 256]; s += v[e]; }
  red[tid] = s; __syncthreads();
  for (int off = 128; off; off >>= 1) {
    if (tid < off) red[tid] += red[tid + off];
    __syncthreads();
  }
  const float mu = red[0] * (1.0f / 1024.0f);
  __syncthreads();
  float vs = 0.f;
#pragma unroll
  for (int e = 0; e < 4; ++e) { const float d = v[e] - mu; vs += d * d; }
  red[tid] = vs; __syncthreads();
  for (int off = 128; off; off >>= 1) {
    if (tid < off) red[tid] += red[tid + off];
    __syncthreads();
  }
  const float rstd = rsqrtf(red[0] * (1.0f / 1024.0f) + 1e-5f);
#pragma unroll
  for (int e = 0; e < 4; ++e) {
    const int c = tid + e * 256;
    out[(size_t)row * 1024 + c] = (v[e] - mu) * rstd * gam[c] + bet[c];
  }
}

// =============================== launch =====================================
extern "C" void kernel_launch(void* const* d_in, const int* in_sizes, int n_in,
                              void* d_out, int out_size, void* d_ws, size_t ws_size,
                              hipStream_t stream) {
  const float* x     = (const float*)d_in[0];
  const float* pos   = (const float*)d_in[1];
  const float* W_qkv = (const float*)d_in[2];
  const float* b_qkv = (const float*)d_in[3];
  const float* W_out = (const float*)d_in[4];
  const float* b_out = (const float*)d_in[5];
  const float* ln_g  = (const float*)d_in[6];
  const float* ln_b  = (const float*)d_in[7];
  float* out = (float*)d_out;

  // workspace layout (bytes)
  char* ws = (char*)d_ws;
  float*  qkv = (float*)(ws + 0);           // 4096*3072*4  = 48 MB
  __bf16* qb  = (__bf16*)(ws + 50331648);   // 2*16*2048*64*2 = 8 MB
  __bf16* kb  = (__bf16*)(ws + 58720256);   // 8 MB
  __bf16* vt  = (__bf16*)(ws + 67108864);   // 8 MB (transposed [64][S])
  float*  ctx = (float*)(ws + 75497472);    // 16 MB
  float*  y   = (float*)(ws + 92274688);    // 16 MB  -> total 104 MB
  if (ws_size < (size_t)109051904) return;

  const dim3 blk(256);
  // 1) qkv = x @ W_qkv + b_qkv            [4096 x 3072]
  gemm_bf16_kernel<3072, false><<<dim3(24, 32), blk, 0, stream>>>(
      x, W_qkv, b_qkv, nullptr, qkv, 1024);
  // 2) k-norm + RoPE + bf16 pack (q,k row-major; v transposed)
  rope_pack_kernel<<<dim3(8192), blk, 0, stream>>>(qkv, pos, qb, kb, vt);
  // 3) flash attention -> ctx             [4096 x 1024]
  attn_kernel<<<dim3(512), blk, 0, stream>>>(qb, kb, vt, ctx);
  // 4) y = ctx @ W_out + b_out + x        [4096 x 1024]
  gemm_bf16_kernel<1024, true><<<dim3(8, 32), blk, 0, stream>>>(
      ctx, W_out, b_out, x, y, 1024);
  // 5) LayerNorm(y) -> out
  ln_kernel<<<dim3(4096), blk, 0, stream>>>(y, ln_g, ln_b, out);
}
